// SlidingCacheAttention_58136677318818
// MI455X (gfx1250) — matrix operand order repaired and checked by reading, not compiled, not run on hardware
//
#include <hip/hip_runtime.h>

// ---------------------------------------------------------------------------
// SlidingCacheAttention for MI455X (gfx1250, wave32, WMMA bf16 16x16x32)
// ---------------------------------------------------------------------------

#define DEVINL __device__ __forceinline__

typedef __bf16 bf16_t;
typedef __attribute__((ext_vector_type(16))) __bf16 bf16x16;
typedef __attribute__((ext_vector_type(8)))  float  f32x8;

constexpr int BATCH  = 4;
constexpr int SEQ    = 1024;
constexpr int DIMSZ  = 2048;
constexpr int NH     = 16;
constexpr int NKV    = 4;
constexpr int HD     = 128;
constexpr int CACHEN = 3072;
constexpr int WIN    = 2048;
constexpr int MTOT   = CACHEN + SEQ;      // 4096
constexpr float EPSF  = 1e-6f;
constexpr float QK_SCALE = 0.08838834764831845f; // 1/sqrt(128)

union PackU { bf16_t h[2]; unsigned int u; };
union ABFrag { unsigned int u[8]; bf16x16 v; };

// pack two consecutive elements (converted to bf16) into one dword
DEVINL unsigned int ldpair(const float* p) {
  PackU c; c.h[0] = (bf16_t)p[0]; c.h[1] = (bf16_t)p[1]; return c.u;
}
DEVINL unsigned int ldpair(const bf16_t* p) {
  return *(const unsigned int*)p;   // 4B-aligned: even bf16 offsets only
}

// A operand (16x32 bf16): lane(0..15)=row M, lane>=16 same rows K+8 half.
// VGPR v<4 : K = 2v,2v+1 (+8*half) ; v>=4 : K = 16+2(v-4) (+8*half)
DEVINL int a_koff(int v, int kh) { return (v < 4 ? 2*v : 16 + 2*(v-4)) + 8*kh; }
// B operand (32x16 bf16): lane&15 = col N, VGPR v: same K as A (v<4: 2v ; v>=4: 16+2(v-4)) (+8*half)
DEVINL int b_koff(int v, int kh) { return (v < 4 ? 2*v : 16 + 2*(v-4)) + 8*kh; }

DEVINL f32x8 wmma_bf16(bf16x16 a, bf16x16 b, f32x8 c) {
  return __builtin_amdgcn_wmma_f32_16x16x32_bf16(
      /*neg_a=*/false, a, /*neg_b=*/false, b,
      /*c_mod=*/(short)0, c, /*reuse_a=*/false, /*reuse_b=*/false);
}

// ---------------------------------------------------------------------------
// past_k -> bf16 K-cache (B,HKV,MTOT,D); past_v -> bf16 V^T (B,HKV,D,MTOT)
// ---------------------------------------------------------------------------
__global__ __launch_bounds__(256) void cvt_cache(const float* __restrict__ pk,
                                                 const float* __restrict__ pv,
                                                 bf16_t* __restrict__ kall,
                                                 bf16_t* __restrict__ vT) {
  size_t total = (size_t)BATCH * NKV * CACHEN * HD;
  for (size_t i = (size_t)blockIdx.x * blockDim.x + threadIdx.x; i < total;
       i += (size_t)gridDim.x * blockDim.x) {
    int d = (int)(i % HD);
    size_t t = i / HD;
    int c  = (int)(t % CACHEN); t /= CACHEN;
    int hh = (int)(t % NKV);
    int b  = (int)(t / NKV);
    kall[(((size_t)(b*NKV+hh)*MTOT) + c)*HD + d] = (bf16_t)pk[i];
    vT  [(((size_t)(b*NKV+hh)*HD)   + d)*MTOT + c] = (bf16_t)pv[i];
  }
}

// ---------------------------------------------------------------------------
// C = A(Mx K) * Bw(N x K)^T  with bf16 WMMA, fp32 accumulate.
// block = 256 thr (8 waves); wave tile = 16 rows x 64 cols; block = 64 x 128.
// MODE 0: bf16 row-major out   MODE 1: scatter -> K cache at CACHE+n
// MODE 2: scatter -> V^T at CACHE+n      MODE 3: fp32 out + bias
// ---------------------------------------------------------------------------
template <int MODE, typename TA, typename TB>
__global__ __launch_bounds__(256) void gemm_wmma(const TA* __restrict__ A,
                                                 const TB* __restrict__ Bw,
                                                 bf16_t* __restrict__ outb,
                                                 float*  __restrict__ outf,
                                                 const float* __restrict__ bias,
                                                 int K, int ldc) {
  const int wave = threadIdx.x >> 5, lane = threadIdx.x & 31;
  const int nl = lane & 15, kh = lane >> 4;
  const int rowBase = blockIdx.x * 64  + (wave & 3) * 16;
  const int colBase = blockIdx.y * 128 + (wave >> 2) * 64;

  f32x8 acc[4] = {};
  for (int k0 = 0; k0 < K; k0 += 32) {
    ABFrag a;
#pragma unroll
    for (int v = 0; v < 8; v++)
      a.u[v] = ldpair(A + (size_t)(rowBase + nl) * K + k0 + a_koff(v, kh));
#pragma unroll
    for (int c = 0; c < 4; c++) {
      ABFrag bb;
#pragma unroll
      for (int v = 0; v < 8; v++)
        bb.u[v] = ldpair(Bw + (size_t)(colBase + c*16 + nl) * K + k0 + b_koff(v, kh));
      acc[c] = wmma_bf16(a.v, bb.v, acc[c]);
    }
  }
#pragma unroll
  for (int c = 0; c < 4; c++) {
#pragma unroll
    for (int r = 0; r < 8; r++) {
      int grow = rowBase + r + 8*kh;          // C layout: vgpr r -> row r+8*half
      int gcol = colBase + c*16 + nl;
      float val = acc[c][r];
      if constexpr (MODE == 0) {
        outb[(size_t)grow * ldc + gcol] = (bf16_t)val;
      } else if constexpr (MODE == 1) {
        int b = grow / SEQ, n = grow % SEQ, hh = gcol / HD, d = gcol % HD;
        outb[(((size_t)(b*NKV+hh)*MTOT) + CACHEN + n)*HD + d] = (bf16_t)val;
      } else if constexpr (MODE == 2) {
        int b = grow / SEQ, n = grow % SEQ, hh = gcol / HD, d = gcol % HD;
        outb[(((size_t)(b*NKV+hh)*HD) + d)*MTOT + CACHEN + n] = (bf16_t)val;
      } else {
        outf[(size_t)grow * ldc + gcol] = val + bias[gcol];
      }
    }
  }
}

// ---------------------------------------------------------------------------
// In-place RMSNorm + RoPE, one wave per 128-element head row.
// mode 0: q rows (B,SEQ,NH,HD);  mode 1: new-k rows inside K cache.
// ---------------------------------------------------------------------------
__global__ __launch_bounds__(256) void normrope(bf16_t* __restrict__ base,
                                                const float* __restrict__ w,
                                                const float* __restrict__ fcos,
                                                const float* __restrict__ fsin,
                                                int mode) {
  const int wave = threadIdx.x >> 5, lane = threadIdx.x & 31;
  const int rid = blockIdx.x * 8 + wave;
  bf16_t* row; int n;
  if (mode == 0) {
    int b = rid / (SEQ*NH); int r2 = rid % (SEQ*NH); n = r2 / NH; int h = r2 % NH;
    row = base + (((size_t)(b*SEQ+n)*NH) + h) * HD;
  } else {
    int b = rid / (SEQ*NKV); int r2 = rid % (SEQ*NKV); n = r2 / NKV; int hh = r2 % NKV;
    row = base + (((size_t)(b*NKV+hh)*MTOT) + CACHEN + n) * HD;
  }
  float e[4];
#pragma unroll
  for (int i = 0; i < 4; i++) e[i] = (float)row[4*lane + i];
  float ss = e[0]*e[0] + e[1]*e[1] + e[2]*e[2] + e[3]*e[3];
#pragma unroll
  for (int off = 16; off >= 1; off >>= 1) ss += __shfl_xor(ss, off, 32);
  float inv = rsqrtf(ss * (1.0f / HD) + EPSF);
#pragma unroll
  for (int i = 0; i < 4; i++) e[i] *= inv * w[4*lane + i];
  // RoPE: pairs (4l,4l+1)->j=2l ; (4l+2,4l+3)->j=2l+1 ; freqs row stride = D/2
  float c0 = fcos[n*(HD/2) + 2*lane    ], s0 = fsin[n*(HD/2) + 2*lane    ];
  float c1 = fcos[n*(HD/2) + 2*lane + 1], s1 = fsin[n*(HD/2) + 2*lane + 1];
  float r0 = e[0]*c0 - e[1]*s0, r1 = e[0]*s0 + e[1]*c0;
  float r2 = e[2]*c1 - e[3]*s1, r3 = e[2]*s1 + e[3]*c1;
  row[4*lane+0] = (bf16_t)r0; row[4*lane+1] = (bf16_t)r1;
  row[4*lane+2] = (bf16_t)r2; row[4*lane+3] = (bf16_t)r3;
}

// ---------------------------------------------------------------------------
// Flash attention, sliding window. One wave = 16 queries, full D=128.
// grid = (SEQ/128, NH, BATCH), block = 256 (8 waves).
// ---------------------------------------------------------------------------
__global__ __launch_bounds__(256) void flash_attn(const bf16_t* __restrict__ qb,
                                                  const bf16_t* __restrict__ kall,
                                                  const bf16_t* __restrict__ vT,
                                                  bf16_t* __restrict__ attnb) {
  __shared__ bf16_t plds[8][16*32];   // per-wave P staging (C-layout -> A-layout)
  const int wave = threadIdx.x >> 5, lane = threadIdx.x & 31;
  const int nl = lane & 15, kh = lane >> 4;
  const int q0  = blockIdx.x * 128 + wave * 16;
  const int h   = blockIdx.y, b = blockIdx.z;
  const int hkv = h / (NH / NKV);

  const bf16_t* Q  = qb   + (((size_t)(b*SEQ + q0)*NH) + h) * HD;
  const bf16_t* Kc = kall + ((size_t)(b*NKV + hkv) * MTOT) * HD;
  const bf16_t* Vt = vT   + ((size_t)(b*NKV + hkv) * HD) * MTOT;

  // preload Q as 4 A-operand fragments (d-steps of 32)
  ABFrag aq[4];
#pragma unroll
  for (int ds = 0; ds < 4; ds++)
#pragma unroll
    for (int v = 0; v < 8; v++)
      aq[ds].u[v] = ldpair(Q + (size_t)nl * (NH*HD) + ds*32 + a_koff(v, kh));

  f32x8 O[8] = {};
  float mrow[8], lrow[8];
#pragma unroll
  for (int r = 0; r < 8; r++) { mrow[r] = -1e30f; lrow[r] = 0.0f; }

  const int kstart = ((CACHEN + q0) - WIN + 1) & ~31;  // window lower edge
  const int klast  = CACHEN + q0 + 15;                 // diagonal edge
  for (int kb = kstart; kb <= klast; kb += 32) {
    // S = Q * K^T for 32 keys -> two 16x16 f32 tiles
    f32x8 S0 = {}, S1 = {};
#pragma unroll
    for (int ds = 0; ds < 4; ds++) {
      ABFrag b0, b1;
#pragma unroll
      for (int v = 0; v < 8; v++) {
        int koff = ds*32 + b_koff(v, kh);
        b0.u[v] = ldpair(Kc + (size_t)(kb      + nl) * HD + koff);
        b1.u[v] = ldpair(Kc + (size_t)(kb + 16 + nl) * HD + koff);
      }
      S0 = wmma_bf16(aq[ds].v, b0.v, S0);
      S1 = wmma_bf16(aq[ds].v, b1.v, S1);
    }
    // mask + online softmax (per query row; row lives across one 16-lane half)
    float sc[8];
#pragma unroll
    for (int r = 0; r < 8; r++) {
      int m = r + 8*kh;
      int qpos = CACHEN + q0 + m;
      int kp0 = kb + nl, kp1 = kb + 16 + nl;
      bool ok0 = (kp0 <= qpos) && (qpos - kp0 < WIN);
      bool ok1 = (kp1 <= qpos) && (qpos - kp1 < WIN);
      float s0 = ok0 ? S0[r] * QK_SCALE : -1e30f;
      float s1 = ok1 ? S1[r] * QK_SCALE : -1e30f;
      float t = fmaxf(s0, s1);
#pragma unroll
      for (int off = 8; off >= 1; off >>= 1) t = fmaxf(t, __shfl_xor(t, off, 16));
      float mnew = fmaxf(mrow[r], t);
      float p0 = __expf(s0 - mnew), p1 = __expf(s1 - mnew);
      float rs = p0 + p1;
#pragma unroll
      for (int off = 8; off >= 1; off >>= 1) rs += __shfl_xor(rs, off, 16);
      sc[r]   = __expf(mrow[r] - mnew);
      lrow[r] = lrow[r] * sc[r] + rs;
      mrow[r] = mnew;
      plds[wave][m*32 +      nl] = (bf16_t)p0;   // C-layout -> LDS row-major
      plds[wave][m*32 + 16 + nl] = (bf16_t)p1;
    }
#pragma unroll
    for (int dt = 0; dt < 8; dt++)
#pragma unroll
      for (int r = 0; r < 8; r++) O[dt][r] *= sc[r];
    // P back as A operand (same wave; LDS in-order, compiler inserts dscnt wait)
    ABFrag ap;
#pragma unroll
    for (int v = 0; v < 8; v++)
      ap.u[v] = *(const unsigned int*)&plds[wave][nl*32 + a_koff(v, kh)];
    // O += P * V   (V^T: consecutive keys contiguous -> dword pair loads)
#pragma unroll
    for (int dt = 0; dt < 8; dt++) {
      ABFrag bv;
#pragma unroll
      for (int v = 0; v < 8; v++)
        bv.u[v] = ldpair(Vt + (size_t)(dt*16 + nl) * MTOT + kb + b_koff(v, kh));
      O[dt] = wmma_bf16(ap.v, bv.v, O[dt]);
    }
  }
  // epilogue: O / l -> bf16 attn (B,SEQ,NH*HD)
#pragma unroll
  for (int dt = 0; dt < 8; dt++) {
#pragma unroll
    for (int r = 0; r < 8; r++) {
      int m = r + 8*kh;
      attnb[(((size_t)(b*SEQ + q0 + m)*NH) + h)*HD + dt*16 + nl] =
          (bf16_t)(O[dt][r] / lrow[r]);
    }
  }
}

// ---------------------------------------------------------------------------
extern "C" void kernel_launch(void* const* d_in, const int* in_sizes, int n_in,
                              void* d_out, int out_size, void* d_ws, size_t ws_size,
                              hipStream_t stream) {
  (void)in_sizes; (void)n_in; (void)out_size; (void)ws_size;
  const float* x      = (const float*)d_in[0];
  const float* past_k = (const float*)d_in[1];
  const float* past_v = (const float*)d_in[2];
  const float* fcos   = (const float*)d_in[3];
  const float* fsin   = (const float*)d_in[4];
  const float* Wq     = (const float*)d_in[5];
  const float* Wk     = (const float*)d_in[6];
  const float* Wv     = (const float*)d_in[7];
  const float* Wo     = (const float*)d_in[8];
  const float* bo     = (const float*)d_in[9];
  const float* wqn    = (const float*)d_in[10];
  const float* wkn    = (const float*)d_in[11];
  float* out = (float*)d_out;

  // workspace: 4 x 16 MiB bf16 buffers (qb, K-cache, V^T, attn)
  char* ws = (char*)d_ws;
  bf16_t* qb    = (bf16_t*)(ws + ((size_t)0  << 20));
  bf16_t* kall  = (bf16_t*)(ws + ((size_t)16 << 20));
  bf16_t* vT    = (bf16_t*)(ws + ((size_t)32 << 20));
  bf16_t* attnb = (bf16_t*)(ws + ((size_t)48 << 20));

  // 1) bf16 cache conversion (K row-major, V transposed)
  cvt_cache<<<4096, 256, 0, stream>>>(past_k, past_v, kall, vT);

  // 2) projections: q (4096x2048), k (4096x512 -> cache), v (4096x512 -> V^T)
  gemm_wmma<0, float, float><<<dim3(64, 16), 256, 0, stream>>>(
      x, Wq, qb, nullptr, nullptr, DIMSZ, NH*HD);
  gemm_wmma<1, float, float><<<dim3(64, 4), 256, 0, stream>>>(
      x, Wk, kall, nullptr, nullptr, DIMSZ, NKV*HD);
  gemm_wmma<2, float, float><<<dim3(64, 4), 256, 0, stream>>>(
      x, Wv, vT, nullptr, nullptr, DIMSZ, NKV*HD);

  // 3) RMSNorm + RoPE in place (q rows, then new-k rows inside the cache)
  normrope<<<(BATCH*SEQ*NH)/8,  256, 0, stream>>>(qb,   wqn, fcos, fsin, 0);
  normrope<<<(BATCH*SEQ*NKV)/8, 256, 0, stream>>>(kall, wkn, fcos, fsin, 1);

  // 4) sliding-window flash attention
  flash_attn<<<dim3(SEQ/128, NH, BATCH), 256, 0, stream>>>(qb, kall, vT, attnb);

  // 5) output projection + bias -> fp32
  gemm_wmma<3, bf16_t, float><<<dim3(64, 16), 256, 0, stream>>>(
      attnb, Wo, nullptr, out, bo, NH*HD, DIMSZ);
}
